// MoCE_78237124263950
// MI455X (gfx1250) — matrix operand order repaired and checked by reading, compiled, not run on hardware
//
#include <hip/hip_runtime.h>
#include <math.h>

// ---------------------------------------------------------------------------
// MoE forward for MI455X (gfx1250, wave32, WMMA bf16 16x16x32, f32 accum)
// B=4096, I=512, H=1024, O=512, E=16, k=4
//
// Strategy: weights are pre-packed (fp32 -> bf16, WMMA B-fragment order) into
// workspace once per launch, so the hot GEMM loops issue pure b128 loads into
// v_wmma with no per-element conversion VALU. Activations (x tile, h tile)
// live in LDS in A-fragment order -> 2 x ds_load_b128 per fragment.
// ---------------------------------------------------------------------------

typedef __attribute__((ext_vector_type(16))) __bf16        v16bf;
typedef __attribute__((ext_vector_type(8)))  float         v8f;
typedef __attribute__((ext_vector_type(4)))  unsigned int  v4u;

union FragAB {
    v16bf        bf;
    unsigned int u[8];
    v4u          q[2];
};

__device__ __forceinline__ unsigned short f2bf(float f) {
    // round-to-nearest-even float -> bf16
    unsigned int u = __float_as_uint(f);
    unsigned int r = (u + 0x7FFFu + ((u >> 16) & 1u)) >> 16;
    return (unsigned short)r;
}
__device__ __forceinline__ unsigned int pack2bf(float lo, float hi) {
    return (unsigned int)f2bf(lo) | ((unsigned int)f2bf(hi) << 16);
}

#define EPS_F 2.220446049250313e-16f

// ---------------------------------------------------------------------------
// Weight pre-pack: fp32 [E][K][N] -> bf16 fragments [e][nblk][kblk][lane][j]
// One dword (two packed bf16) per thread-iteration.
//   B-fragment layout (CDNA5 ISA 7.12.2, 16-bit B 32x16):
//     lane: col = lane&15; dword j holds rows kr,kr+1, kr = (j<4?0:16)+(lane>>4)*8+(j&3)*2.
// ---------------------------------------------------------------------------
__global__ __launch_bounds__(256)
void pack_w1_kernel(const float* __restrict__ W, unsigned int* __restrict__ P) {
    // W1: K=512 (16 kblk), N=1024 (64 nblk), stride N=1024
    const unsigned int id = blockIdx.x * 256 + threadIdx.x;   // 4,194,304 total
    const int j    = id & 7;
    const int lane = (id >> 3) & 31;
    const int kb   = (id >> 8) & 15;
    const int nb   = (id >> 12) & 63;
    const int e    = id >> 18;
    const int col  = lane & 15;
    const int krow = kb * 32 + ((j < 4) ? 0 : 16) + (lane >> 4) * 8 + (j & 3) * 2;
    const size_t base = (size_t)e * 512 * 1024 + (size_t)krow * 1024 + nb * 16 + col;
    P[id] = pack2bf(W[base], W[base + 1024]);
}

__global__ __launch_bounds__(256)
void pack_w2_kernel(const float* __restrict__ W, unsigned int* __restrict__ P) {
    // W2: K=1024 (32 kblk), N=512 (32 nblk), stride N=512
    const unsigned int id = blockIdx.x * 256 + threadIdx.x;   // 4,194,304 total
    const int j    = id & 7;
    const int lane = (id >> 3) & 31;
    const int kb   = (id >> 8) & 31;
    const int nb   = (id >> 13) & 31;
    const int e    = id >> 18;
    const int col  = lane & 15;
    const int krow = kb * 32 + ((j < 4) ? 0 : 16) + (lane >> 4) * 8 + (j & 3) * 2;
    const size_t base = (size_t)e * 1024 * 512 + (size_t)krow * 512 + nb * 16 + col;
    P[id] = pack2bf(W[base], W[base + 512]);
}

// ---------------------------------------------------------------------------
// Gating: logits = x @ w_gate; top-4 softmax scattered to dense [B,16]
// One wave32 per token. Lanes 0..15 each own one expert column.
// ---------------------------------------------------------------------------
__global__ __launch_bounds__(256)
void moe_gate_kernel(const float* __restrict__ x,
                     const float* __restrict__ wg,
                     float* __restrict__ gates) {
    const int token = blockIdx.x * (blockDim.x >> 5) + (threadIdx.x >> 5);
    const int lane  = threadIdx.x & 31;

    float acc = 0.f;
    if (lane < 16) {
        const float* xr = x + (size_t)token * 512;
        for (int i = 0; i < 512; ++i)
            acc = fmaf(xr[i], wg[i * 16 + lane], acc);
    }

    float lg[16];
#pragma unroll
    for (int e = 0; e < 16; ++e) lg[e] = __shfl(acc, e, 32);

    if (lane == 0) {
        float work[16], g[16];
#pragma unroll
        for (int e = 0; e < 16; ++e) { work[e] = lg[e]; g[e] = 0.f; }
        int   idx[4];
        float tv[4];
        for (int j = 0; j < 4; ++j) {          // serial top-4 (first-max ties)
            int bi = 0; float bv = work[0];
            for (int e = 1; e < 16; ++e)
                if (work[e] > bv) { bv = work[e]; bi = e; }
            idx[j] = bi; tv[j] = bv; work[bi] = -3.4e38f;
        }
        const float mx = tv[0];
        float s = 0.f, ex[4];
        for (int j = 0; j < 4; ++j) { ex[j] = expf(tv[j] - mx); s += ex[j]; }
        for (int j = 0; j < 4; ++j) g[idx[j]] = ex[j] / s;
        for (int e = 0; e < 16; ++e) gates[(size_t)token * 16 + e] = g[e];
    }
}

// ---------------------------------------------------------------------------
// Fused expert MLP + combine. One block = 16 tokens, 8 waves.
// LDS x/h tiles live in A-fragment order: [kblk][lane][8 dwords] (1 KB/kblk),
// so a fragment = two ds_load_b128. Packed weights = two global_load_b128.
// ---------------------------------------------------------------------------
__global__ __launch_bounds__(256)
void moe_expert_kernel(const float* __restrict__ x,
                       const unsigned int* __restrict__ W1p, // packed bf16 frags
                       const float* __restrict__ b1,
                       const unsigned int* __restrict__ W2p, // packed bf16 frags
                       const float* __restrict__ b2,
                       const float* __restrict__ gates,
                       float* __restrict__ out) {
    __shared__ v4u   lx4[16 * 64];   // x tile fragments: 16 kblk x 1KB = 16 KB
    __shared__ v4u   lh4[32 * 64];   // h tile fragments: 32 kblk x 1KB = 32 KB
    __shared__ float lgate[16];

    const int tid  = threadIdx.x;
    const int lane = tid & 31;
    const int w    = tid >> 5;        // wave id 0..7
    const int nlo  = lane & 15;       // fragment column
    const int hsel = lane >> 4;       // K-half select
    const int row0 = blockIdx.x * 16; // first token of this tile

    // ---- stage x tile straight into A-fragment order (fp32 -> bf16) ------
    // A layout (16-bit 16x32): lane&15 = row; dword j, K-half h:
    //   ka = (j<4 ? 0 : 16) + h*8 + (j&3)*2
    for (int i = tid; i < 16 * 256; i += 256) {
        const int j  = i & 7;
        const int ln = (i >> 3) & 31;
        const int kb = i >> 8;
        const int rw = ln & 15;
        const int ka = ((j < 4) ? 0 : 16) + (ln >> 4) * 8 + (j & 3) * 2;
        const size_t g = (size_t)(row0 + rw) * 512 + kb * 32 + ka;
        ((unsigned int*)lx4)[i] = pack2bf(x[g], x[g + 1]);
    }

    float comb[4][8];
#pragma unroll
    for (int t = 0; t < 4; ++t)
#pragma unroll
        for (int r = 0; r < 8; ++r) comb[t][r] = 0.f;

    // precomputed inverse fragment coords for the fc1-output column this lane
    // owns in its C fragments (n varies only with nt by +16 -> same kk path
    // per nt; compute inside the nt loop instead).

    for (int e = 0; e < 16; ++e) {
        __syncthreads();                      // lh free, prev gate reads done
        if (tid < 16) lgate[tid] = gates[(size_t)(row0 + tid) * 16 + e];

        const float* b1e = b1 + e * 1024;

        // prefetch next expert's packed W1 into L2 (global_prefetch_b8)
        if (e + 1 < 16)
            __builtin_prefetch(W1p + (size_t)(e + 1) * 64 * 16 * 256 + tid * 1024, 0, 1);

        // ---------------- fc1: h = relu(x @ W1[e] + b1[e]) ----------------
        for (int nt = 0; nt < 8; ++nt) {
            const int nb = w * 8 + nt;
            const int n0 = nb * 16;
            const float bv = b1e[n0 + nlo];
            v8f c;
#pragma unroll
            for (int r = 0; r < 8; ++r) c[r] = bv;

            const v4u* bp = (const v4u*)W1p + ((size_t)(e * 64 + nb) * 16) * 64 + lane * 2;
            for (int kb = 0; kb < 16; ++kb) {
                FragAB a, b;
                a.q[0] = lx4[kb * 64 + lane * 2];
                a.q[1] = lx4[kb * 64 + lane * 2 + 1];
                b.q[0] = bp[kb * 64];
                b.q[1] = bp[kb * 64 + 1];
                c = __builtin_amdgcn_wmma_f32_16x16x32_bf16(
                        false, a.bf, false, b.bf, (short)0, c, false, false);
            }
            // relu + bf16 store into lh fragment layout.
            // value h[m][n]: n fixed per lane; invert kk=n&31 -> (hs,j,b):
            const int n   = n0 + nlo;
            const int kkb = n >> 5;
            const int kk  = n & 31;
            const int hs  = (kk >> 3) & 1;
            const int jj  = ((kk >> 4) << 2) + ((kk & 7) >> 1);
            const int bb  = kk & 1;
            unsigned short* lhs = (unsigned short*)lh4;
#pragma unroll
            for (int r = 0; r < 8; ++r) {
                const int m  = r + 8 * hsel;          // row of this C element
                const int ln = m + 16 * hs;           // reading lane
                lhs[kkb * 512 + ln * 16 + jj * 2 + bb] = f2bf(fmaxf(c[r], 0.f));
            }
        }
        __syncthreads();                      // lh complete, lgate visible

        // ------- fc2 + epilogue: comb += g * exp(tanh(h@W2+b2)*10) --------
        const float* b2e = b2 + e * 512;
        for (int nt = 0; nt < 4; ++nt) {
            const int nb = w * 4 + nt;
            const int n0 = nb * 16;
            const float bv = b2e[n0 + nlo];
            v8f c;
#pragma unroll
            for (int r = 0; r < 8; ++r) c[r] = bv;

            const v4u* bp = (const v4u*)W2p + ((size_t)(e * 32 + nb) * 32) * 64 + lane * 2;
            for (int kb = 0; kb < 32; ++kb) {
                FragAB a, b;
                a.q[0] = lh4[kb * 64 + lane * 2];
                a.q[1] = lh4[kb * 64 + lane * 2 + 1];
                b.q[0] = bp[kb * 64];
                b.q[1] = bp[kb * 64 + 1];
                c = __builtin_amdgcn_wmma_f32_16x16x32_bf16(
                        false, a.bf, false, b.bf, (short)0, c, false, false);
            }
#pragma unroll
            for (int r = 0; r < 8; ++r) {
                const int   m = r + 8 * hsel;
                const float d = tanhf(c[r]) * 10.f;
                comb[nt][r] = fmaf(lgate[m], expf(d), comb[nt][r]);
            }
        }
    }

    // -------------------- final: log(max(c, EPS)) -> out ------------------
#pragma unroll
    for (int nt = 0; nt < 4; ++nt) {
        const int n0 = (w * 4 + nt) * 16;
#pragma unroll
        for (int r = 0; r < 8; ++r) {
            const int m = r + 8 * hsel;
            float v = comb[nt][r];
            v = (v == 0.f) ? EPS_F : v;
            out[(size_t)(row0 + m) * 512 + n0 + nlo] = logf(v);
        }
    }
}

// ---------------------------------------------------------------------------
extern "C" void kernel_launch(void* const* d_in, const int* in_sizes, int n_in,
                              void* d_out, int out_size, void* d_ws, size_t ws_size,
                              hipStream_t stream) {
    (void)in_sizes; (void)n_in; (void)out_size; (void)ws_size;
    const float* x  = (const float*)d_in[0];
    const float* wg = (const float*)d_in[1];
    const float* W1 = (const float*)d_in[2];
    const float* b1 = (const float*)d_in[3];
    const float* W2 = (const float*)d_in[4];
    const float* b2 = (const float*)d_in[5];
    // d_in[6] is k (==4), baked into the kernels.

    // workspace layout (16B-aligned slices):
    //   [0, 256KB)              gates fp32 [4096][16]
    //   [256KB, 256KB+16MB)     W1 packed bf16 fragments
    //   [+16MB, +32MB)          W2 packed bf16 fragments
    char* ws = (char*)d_ws;
    float*        gates = (float*)ws;
    unsigned int* W1p   = (unsigned int*)(ws + (size_t)262144);
    unsigned int* W2p   = (unsigned int*)(ws + (size_t)262144 + (16u << 20));
    float* out = (float*)d_out;

    pack_w1_kernel<<<16384, 256, 0, stream>>>(W1, W1p);
    pack_w2_kernel<<<16384, 256, 0, stream>>>(W2, W2p);
    moe_gate_kernel<<<512, 256, 0, stream>>>(x, wg, gates);
    moe_expert_kernel<<<256, 256, 0, stream>>>(x, W1p, b1, W2p, b2, gates, out);
}
